// LSTMLayer_28845000360565
// MI455X (gfx1250) — matrix operand order repaired and checked by reading, not compiled, not run on hardware
//
#include <hip/hip_runtime.h>
#include <math.h>

// ---------------------------------------------------------------------------
// BatchNorm + Bidirectional LSTM (return last hidden, concat) for MI455X.
// B=32, T=512, C=512, H=512.  All GEMMs via v_wmma_f32_16x16x32_bf16.
// Input-projection GEMM stages B-tiles through LDS with
// global_load_async_to_lds_b128 (ASYNCcnt) + double buffering.
// ---------------------------------------------------------------------------

#define Bsz 32
#define Tsz 512
#define Csz 512
#define Hsz 512
#define Gsz 2048   // 4*H

typedef __attribute__((ext_vector_type(16))) __bf16 bf16x16;
typedef __attribute__((ext_vector_type(8)))  __bf16 bf16x8;
typedef __attribute__((ext_vector_type(8)))  float  f32x8;

// ---- WMMA fragment helpers (wave32; A/B use contiguous k = 16*(lane>>4) + elem) ----------

// A fragment: 16x32 bf16.  lane m = lane&15; elems 0..15 -> K=kb+kh..kb+kh+15
// (kh = (lane>>4)*16)
__device__ __forceinline__ bf16x16 load_a_frag(const __bf16* row, int lane, int kb) {
  const int kh = (lane >> 4) * 16;
  bf16x8 lo = *reinterpret_cast<const bf16x8*>(row + kb + kh);
  bf16x8 hi = *reinterpret_cast<const bf16x8*>(row + kb + kh + 8);
  bf16x16 r;
#pragma unroll
  for (int i = 0; i < 8; ++i) { r[i] = lo[i]; r[i + 8] = hi[i]; }
  return r;
}

// B fragment: 32x16 bf16 from W^T stored [N][K] (K contiguous).
// lane n = lane&15, lanes 16..31 hold K=16..31 -> one contiguous 32B load.
__device__ __forceinline__ bf16x16 load_b_frag(const __bf16* BT, int ldk,
                                               int nbase, int kb, int lane) {
  const int n  = lane & 15;
  const int kh = (lane >> 4) * 16;
  return *reinterpret_cast<const bf16x16*>(BT + (long)(nbase + n) * ldk + kb + kh);
}

// C/D fragment: 16x16 f32. lane n = lane&15; VGPR q -> M = (lane>>4)*8 + q.
__device__ __forceinline__ f32x8 load_c_frag(const float* Cp, int lane, int ldc) {
  const int n  = lane & 15;
  const int mh = (lane >> 4) * 8;
  f32x8 r;
#pragma unroll
  for (int q = 0; q < 8; ++q) r[q] = Cp[(long)(mh + q) * ldc + n];
  return r;
}
__device__ __forceinline__ void store_d_frag(float* Cp, int lane, int ldc, f32x8 d) {
  const int n  = lane & 15;
  const int mh = (lane >> 4) * 8;
#pragma unroll
  for (int q = 0; q < 8; ++q) Cp[(long)(mh + q) * ldc + n] = d[q];
}

__device__ __forceinline__ float sigmoidf_(float x) { return 1.0f / (1.0f + expf(-x)); }

// Async global -> LDS copy, 16 bytes per active lane (ASYNCcnt tracked).
__device__ __forceinline__ void async_b128_to_lds(unsigned lds_off, const void* gptr) {
  asm volatile("global_load_async_to_lds_b128 %0, %1, off"
               :: "v"(lds_off), "v"(gptr)
               : "memory");
}

// ---- Kernel 1: BatchNorm (inference) + f32 -> bf16 --------------------------
__global__ __launch_bounds__(256) void bn_to_bf16(
    const float* __restrict__ x, const float* __restrict__ gamma,
    const float* __restrict__ beta, const float* __restrict__ mean,
    const float* __restrict__ var, __bf16* __restrict__ xn) {
  int idx = blockIdx.x * 256 + threadIdx.x;     // B*T*C = 8388608 exact
  int c = idx & (Csz - 1);
  float inv = rsqrtf(var[c] + 1e-3f);
  float v = (x[idx] - mean[c]) * inv * gamma[c] + beta[c];
  xn[idx] = (__bf16)v;
}

// ---- Kernel 2: weight transpose + convert: W[K=512][N=2048] -> WT[N][K] bf16
__global__ __launch_bounds__(256) void transpose_w(
    const float* __restrict__ W, __bf16* __restrict__ WT) {
  int idx = blockIdx.x * 256 + threadIdx.x;     // G*C = 1048576 exact
  int n = idx >> 9;
  int k = idx & (Csz - 1);
  WT[idx] = (__bf16)W[(long)k * Gsz + n];
}

// ---- Kernel 3: input projection GEMM: Z[t][b][:] = xn(b, t|rev) . W + bias --
// grid (128, 32, 2): x = 128-row M-block (8 waves x 16 rows), y = 64-col
// N-group shared by the whole block, z = direction.
// B tile (64x32 bf16 = 4 KB) is async-copied into LDS, double buffered.
__global__ __launch_bounds__(256) void proj_gemm(
    const __bf16* __restrict__ xn,
    const __bf16* __restrict__ WTf, const __bf16* __restrict__ WTb,
    const float* __restrict__ biasf, const float* __restrict__ biasb,
    float* __restrict__ Zf, float* __restrict__ Zb) {
  __shared__ __bf16 Bs[2][64 * 32];              // 2 x 4 KB double buffer

  const int dir = blockIdx.z;
  const __bf16* WT   = dir ? WTb : WTf;
  const float*  bias = dir ? biasb : biasf;
  float*        Z    = dir ? Zb : Zf;

  const int tid   = threadIdx.x;
  const int lane  = tid & 31;
  const int wave  = tid >> 5;
  const int nbase = blockIdx.y * 64;
  const int mbase = blockIdx.x * 128 + wave * 16;

  // output row r = t*32 + b  ->  source row of xn is (b, t) or (b, T-1-t)
  const int r = mbase + (lane & 15);
  const int t = r >> 5;
  const int b = r & 31;
  const int srct = dir ? (Tsz - 1 - t) : t;
  const __bf16* arow = xn + ((long)b * Tsz + srct) * Csz;

  // async B-tile fetch mapping: thread tid moves 16 B: row bn, 8 elems at bk
  const int bn = tid >> 2;                       // 0..63
  const int bk = (tid & 3) * 8;                  // 0,8,16,24
  const __bf16* gB = WT + (long)(nbase + bn) * Csz + bk;
  const unsigned wr0 =
      (unsigned)(uintptr_t)(&Bs[0][0]) + (unsigned)(bn * 32 + bk) * 2u;

  f32x8 acc[4];
#pragma unroll
  for (int j = 0; j < 4; ++j) {
    float bv = bias[nbase + j * 16 + (lane & 15)];
#pragma unroll
    for (int q = 0; q < 8; ++q) acc[j][q] = bv;
  }

  const int NCH = Csz / 32;                      // 16 K-chunks
  async_b128_to_lds(wr0, gB);                    // prologue: chunk 0
  for (int c = 0; c < NCH; ++c) {
    if (c + 1 < NCH) {
      async_b128_to_lds(wr0 + (unsigned)((c + 1) & 1) * 4096u,
                        gB + (c + 1) * 32);
      asm volatile("s_wait_asynccnt 0x1" ::: "memory");
    } else {
      asm volatile("s_wait_asynccnt 0x0" ::: "memory");
    }
    __syncthreads();                             // chunk c resident in LDS

    const __bf16* Bbuf = &Bs[c & 1][0];
    bf16x16 a = load_a_frag(arow, lane, c * 32);
#pragma unroll
    for (int j = 0; j < 4; ++j) {
      bf16x16 bm = *reinterpret_cast<const bf16x16*>(
          Bbuf + (j * 16 + (lane & 15)) * 32 + (lane >> 4) * 16);
      acc[j] = __builtin_amdgcn_wmma_f32_16x16x32_bf16(
          false, a, false, bm, (short)0, acc[j], false, false);
    }
    __syncthreads();                             // done reading buffer c&1
  }

  float* Zt = Z + (long)mbase * Gsz + nbase;
#pragma unroll
  for (int j = 0; j < 4; ++j) store_d_frag(Zt + j * 16, lane, Gsz, acc[j]);
}

// ---- Kernel 4: zero LSTM state ---------------------------------------------
__global__ __launch_bounds__(256) void init_state(
    float* __restrict__ hf, float* __restrict__ cf, __bf16* __restrict__ hb) {
  int idx = blockIdx.x * 256 + threadIdx.x;     // 2*B*H = 32768 exact
  hf[idx] = 0.0f; cf[idx] = 0.0f; hb[idx] = (__bf16)0.0f;
}

// ---- Kernel 5: recurrent GEMM for one step: zbuf = Z[t] + h . U -------------
// grid (16, 2): x*4+wave = tile id (2 M-tiles x 32 n-groups), y = direction.
// Latency-bound sequential step: direct global loads (U stays hot in L2).
__global__ __launch_bounds__(128) void rec_gemm(
    const __bf16* __restrict__ hbf,                    // [2][32][512]
    const __bf16* __restrict__ UTf, const __bf16* __restrict__ UTb,
    const float* __restrict__ Zf, const float* __restrict__ Zb,
    float* __restrict__ zbuf,                          // [2][32][2048]
    int t) {
  const int dir = blockIdx.y;
  const __bf16* UT = dir ? UTb : UTf;
  const float* Z   = (dir ? Zb : Zf) + (long)t * (Bsz * Gsz);

  const int lane  = threadIdx.x & 31;
  const int wave  = threadIdx.x >> 5;
  const int tile  = blockIdx.x * 4 + wave;   // 0..63
  const int mbase = (tile >> 5) * 16;
  const int nbase = (tile & 31) * 64;

  const __bf16* arow =
      hbf + (long)dir * Bsz * Hsz + (long)(mbase + (lane & 15)) * Hsz;

  // accumulator starts from the precomputed input projection (incl. bias)
  const float* Zt = Z + (long)mbase * Gsz + nbase;
  f32x8 acc[4];
#pragma unroll
  for (int j = 0; j < 4; ++j) acc[j] = load_c_frag(Zt + j * 16, lane, Gsz);

  for (int kb = 0; kb < Hsz; kb += 32) {
    bf16x16 a = load_a_frag(arow, lane, kb);
#pragma unroll
    for (int j = 0; j < 4; ++j) {
      bf16x16 bm = load_b_frag(UT, Hsz, nbase + j * 16, kb, lane);
      acc[j] = __builtin_amdgcn_wmma_f32_16x16x32_bf16(
          false, a, false, bm, (short)0, acc[j], false, false);
    }
  }

  float* zo = zbuf + (long)dir * Bsz * Gsz + (long)mbase * Gsz + nbase;
#pragma unroll
  for (int j = 0; j < 4; ++j) store_d_frag(zo + j * 16, lane, Gsz, acc[j]);
}

// ---- Kernel 6: gates + state update (fused i,f,g,o) -------------------------
__global__ __launch_bounds__(256) void gate_step(
    const float* __restrict__ zbuf, float* __restrict__ hf,
    float* __restrict__ cf, __bf16* __restrict__ hb) {
  int idx = blockIdx.x * 256 + threadIdx.x;     // 2*B*H = 32768 exact
  int dir = idx >> 14;
  int rem = idx & 16383;
  int b = rem >> 9;
  int h = rem & (Hsz - 1);
  const float* z = zbuf + (long)dir * Bsz * Gsz + (long)b * Gsz;
  float gi = sigmoidf_(z[h]);
  float gf = sigmoidf_(z[Hsz + h]);
  float gg = tanhf(z[2 * Hsz + h]);
  float go = sigmoidf_(z[3 * Hsz + h]);
  float c = gf * cf[idx] + gi * gg;
  float hn = go * tanhf(c);
  cf[idx] = c;
  hf[idx] = hn;
  hb[idx] = (__bf16)hn;
}

// ---- Kernel 7: final concat [B, 2H] ----------------------------------------
__global__ __launch_bounds__(256) void final_copy(
    const float* __restrict__ hf, float* __restrict__ out) {
  int idx = blockIdx.x * 256 + threadIdx.x;     // B*2H = 32768 exact
  int b = idx >> 10;
  int col = idx & 1023;
  int dir = col >> 9;
  int h = col & (Hsz - 1);
  out[idx] = hf[(long)dir * Bsz * Hsz + (long)b * Hsz + h];
}

// ---------------------------------------------------------------------------
extern "C" void kernel_launch(void* const* d_in, const int* in_sizes, int n_in,
                              void* d_out, int out_size, void* d_ws, size_t ws_size,
                              hipStream_t stream) {
  const float* x     = (const float*)d_in[0];
  const float* gamma = (const float*)d_in[1];
  const float* beta  = (const float*)d_in[2];
  const float* mean  = (const float*)d_in[3];
  const float* var   = (const float*)d_in[4];
  const float* Wf    = (const float*)d_in[5];
  const float* Uf    = (const float*)d_in[6];
  const float* bf    = (const float*)d_in[7];
  const float* Wb    = (const float*)d_in[8];
  const float* Ub    = (const float*)d_in[9];
  const float* bb    = (const float*)d_in[10];
  float* out = (float*)d_out;

  unsigned char* ws = (unsigned char*)d_ws;
  size_t off = 0;
  __bf16* xn  = (__bf16*)(ws + off); off += (size_t)Bsz * Tsz * Csz * 2;   // 16 MB
  __bf16* WfT = (__bf16*)(ws + off); off += (size_t)Gsz * Csz * 2;         // 2 MB
  __bf16* WbT = (__bf16*)(ws + off); off += (size_t)Gsz * Csz * 2;
  __bf16* UfT = (__bf16*)(ws + off); off += (size_t)Gsz * Hsz * 2;
  __bf16* UbT = (__bf16*)(ws + off); off += (size_t)Gsz * Hsz * 2;
  float*  Zf  = (float*)(ws + off);  off += (size_t)Tsz * Bsz * Gsz * 4;   // 128 MB
  float*  Zb  = (float*)(ws + off);  off += (size_t)Tsz * Bsz * Gsz * 4;   // 128 MB
  float*  zbf = (float*)(ws + off);  off += (size_t)2 * Bsz * Gsz * 4;     // 512 KB
  float*  hf  = (float*)(ws + off);  off += (size_t)2 * Bsz * Hsz * 4;
  float*  cf  = (float*)(ws + off);  off += (size_t)2 * Bsz * Hsz * 4;
  __bf16* hb  = (__bf16*)(ws + off); off += (size_t)2 * Bsz * Hsz * 2;

  // 1) BatchNorm -> bf16 activations
  bn_to_bf16<<<(Bsz * Tsz * Csz) / 256, 256, 0, stream>>>(x, gamma, beta, mean, var, xn);

  // 2) Transpose + convert all four weight matrices to [N][K] bf16
  transpose_w<<<(Gsz * Csz) / 256, 256, 0, stream>>>(Wf, WfT);
  transpose_w<<<(Gsz * Csz) / 256, 256, 0, stream>>>(Wb, WbT);
  transpose_w<<<(Gsz * Hsz) / 256, 256, 0, stream>>>(Uf, UfT);
  transpose_w<<<(Gsz * Hsz) / 256, 256, 0, stream>>>(Ub, UbT);

  // 3) Hoisted input projections (big WMMA GEMM, both directions)
  proj_gemm<<<dim3((Tsz * Bsz) / 128, Gsz / 64, 2), 256, 0, stream>>>(
      xn, WfT, WbT, bf, bb, Zf, Zb);

  // 4) Zero initial LSTM state
  init_state<<<(2 * Bsz * Hsz) / 256, 256, 0, stream>>>(hf, cf, hb);

  // 5) Sequential recurrence: per-step WMMA GEMM + fused gate update
  for (int t = 0; t < Tsz; ++t) {
    rec_gemm<<<dim3(16, 2), 128, 0, stream>>>(hb, UfT, UbT, Zf, Zb, zbf, t);
    gate_step<<<(2 * Bsz * Hsz) / 256, 256, 0, stream>>>(zbf, hf, cf, hb);
  }

  // 6) Concatenate last hidden states [B, 2H]
  final_copy<<<(Bsz * 2 * Hsz) / 256, 256, 0, stream>>>(hf, out);
  (void)in_sizes; (void)n_in; (void)out_size; (void)ws_size;
}
